// GLABlock_14379550507694
// MI455X (gfx1250) — matrix operand order repaired and checked by reading, not compiled, not run on hardware
//
#include <hip/hip_runtime.h>

// ---------------------------------------------------------------------------
// Types for CDNA5 WMMA (wave32, 16x16x32 bf16 -> f32) and TDM descriptors
// ---------------------------------------------------------------------------
typedef __bf16 bf16;
typedef __attribute__((ext_vector_type(16))) __bf16 v16bf;
typedef __attribute__((ext_vector_type(8)))  __bf16 v8bf;
typedef __attribute__((ext_vector_type(8)))  float  v8f;
typedef __attribute__((ext_vector_type(4)))  unsigned int v4u;
typedef __attribute__((ext_vector_type(8)))  int v8i;
typedef __attribute__((ext_vector_type(4)))  int v4i;

#if defined(__gfx1250__) && __has_builtin(__builtin_amdgcn_tensor_load_to_lds) && \
    __has_builtin(__builtin_amdgcn_s_wait_tensorcnt)
#define GLA_USE_TDM 1
#else
#define GLA_USE_TDM 0
#endif

__device__ __forceinline__ v8f v8f_zero() {
  v8f r;
#pragma unroll
  for (int i = 0; i < 8; ++i) r[i] = 0.0f;
  return r;
}

__device__ __forceinline__ v8f wmma_bf16(v16bf a, v16bf b, v8f c) {
  return __builtin_amdgcn_wmma_f32_16x16x32_bf16(false, a, false, b, (short)0, c,
                                                 false, false);
}

// A-fragment (16x32 bf16, row-major source, contiguous rows):
// lane L: row = L&15 ; elements K = {kb..kb+15}, kb = (L>=16)?16:0
__device__ __forceinline__ v16bf make_afrag(const bf16* rowbase, int kb) {
  v8bf lo = *(const v8bf*)(rowbase + kb);
  v8bf hi = *(const v8bf*)(rowbase + kb + 8);
  return __builtin_shufflevector(lo, hi, 0, 1, 2, 3, 4, 5, 6, 7, 8, 9, 10, 11,
                                 12, 13, 14, 15);
}

// B-fragment (32x16 bf16) gathered from an LDS row-major [K x N] tile:
// lane L: col = L&15 ; e_i = B[khalf + i][col], khalf = base + ((L>=16)?16:0)
__device__ __forceinline__ v16bf lds_bfrag(const bf16* base, int ld, int khalf,
                                           int col) {
  v16bf r;
#pragma unroll
  for (int i = 0; i < 16; ++i) r[i] = base[(khalf + i) * ld + col];
  return r;
}

// A-fragment of a transposed LDS matrix: A[m][k] = base[k*ld + m]
__device__ __forceinline__ v16bf lds_afragT(const bf16* base, int ld, int kb,
                                            int m) {
  v16bf r;
#pragma unroll
  for (int i = 0; i < 8; ++i) r[i] = base[(kb + i) * ld + m];
#pragma unroll
  for (int i = 0; i < 8; ++i) r[8 + i] = base[(kb + 8 + i) * ld + m];
  return r;
}

// ---------------------------------------------------------------------------
// Problem constants
// ---------------------------------------------------------------------------
constexpr int BB = 2, TT = 2048, HM = 1024, HEADS = 16, DK = 32, DV = 64;
constexpr int HK = HEADS * DK;   // 512
constexpr int HV = HEADS * DV;   // 1024
constexpr int BT = BB * TT;      // 4096 tokens
constexpr int D2 = 2 * HM;       // 2048
constexpr int CH = 64;           // GLA chunk length
constexpr int NCH = TT / CH;     // 32 chunks

#if GLA_USE_TDM
// Tensor Data Mover: DMA one [CH x DV] bf16 tile (row stride HV elements) from
// global memory into LDS.  D# layout per CDNA5 ISA ch.8 (2D tensor, groups 2/3
// zero).  TENSORcnt tracks completion.  6-arg builtin form (clang-23).
__device__ __forceinline__ void tdm_load_v_tile(const bf16* gsrc, bf16* ldst) {
  unsigned long long ga = (unsigned long long)gsrc;
  unsigned int la = (unsigned int)(unsigned long long)ldst;  // LDS byte address
  v4u g0;
  g0[0] = 1u;                        // count=1 (valid), user mode, no gather
  g0[1] = la;                        // lds_addr
  g0[2] = (unsigned int)ga;          // global_addr[31:0]
  g0[3] = (unsigned int)((ga >> 32) & 0x01FFFFFFu) | (2u << 30);  // [56:32]+type=2
  v8i g1;
  g1[0] = (int)(1u << 16);           // workgroup_mask=0, data_size=1 (2 bytes)
  g1[1] = (int)((unsigned)HV << 16); // tensor_dim0[15:0]=1024 @ bits[63:48]
  g1[2] = (int)((unsigned)BT << 16); // dim0[31:16]=0 | tensor_dim1[15:0]=4096
  g1[3] = (int)((unsigned)DV << 16); // dim1[31:16]=0 | tile_dim0=64
  g1[4] = (int)CH;                   // tile_dim1=64, tile_dim2=0
  g1[5] = (int)HV;                   // tensor_dim0_stride[31:0]=1024
  g1[6] = 0;                         // stride[47:32]=0, tensor_dim1_stride lo=0
  g1[7] = 0;
  v4i z4;
  z4[0] = 0; z4[1] = 0; z4[2] = 0; z4[3] = 0;
  v8i z8;
#pragma unroll
  for (int i = 0; i < 8; ++i) z8[i] = 0;
  __builtin_amdgcn_tensor_load_to_lds(g0, g1, z4, z4, z8, 0);
}
#endif

// ---------------------------------------------------------------------------
// Weight transpose + f32->bf16 convert:  W[K,N] f32 -> Wt[N,K] bf16
// ---------------------------------------------------------------------------
__global__ __launch_bounds__(256) void transpose_bf16_kernel(
    const float* __restrict__ W, bf16* __restrict__ Wt, int K, int N) {
  int idx = blockIdx.x * 256 + threadIdx.x;
  if (idx >= K * N) return;
  int kk = idx / N, nn = idx % N;
  Wt[(size_t)nn * K + kk] = (bf16)W[idx];
}

// ---------------------------------------------------------------------------
// LayerNorm over HM=1024, writes bf16
// ---------------------------------------------------------------------------
__global__ __launch_bounds__(256) void ln_bf16_kernel(
    const float* __restrict__ x, const float* __restrict__ w,
    const float* __restrict__ b, bf16* __restrict__ h) {
  __shared__ float sred[18];
  const int t = blockIdx.x;
  const float* xr = x + (size_t)t * HM;
  float s = 0.f, ss = 0.f;
  for (int i = threadIdx.x; i < HM; i += 256) {
    float v = xr[i];
    s += v;
    ss += v * v;
  }
#pragma unroll
  for (int off = 16; off; off >>= 1) {
    s += __shfl_down(s, off);
    ss += __shfl_down(ss, off);
  }
  const int wid = threadIdx.x >> 5;
  if ((threadIdx.x & 31) == 0) {
    sred[wid] = s;
    sred[8 + wid] = ss;
  }
  __syncthreads();
  if (threadIdx.x == 0) {
    float a = 0.f, q = 0.f;
    for (int i = 0; i < 8; ++i) {
      a += sred[i];
      q += sred[8 + i];
    }
    float mean = a * (1.0f / HM);
    float var = q * (1.0f / HM) - mean * mean;
    sred[16] = mean;
    sred[17] = rsqrtf(var + 1e-5f);
  }
  __syncthreads();
  const float mean = sred[16], inv = sred[17];
  for (int i = threadIdx.x; i < HM; i += 256)
    h[(size_t)t * HM + i] = (bf16)((xr[i] - mean) * inv * w[i] + b[i]);
}

// ---------------------------------------------------------------------------
// low = h @ gk_w1   [BT,1024] x [1024,16]
// ---------------------------------------------------------------------------
__global__ __launch_bounds__(256) void lowrank_kernel(
    const bf16* __restrict__ h, const float* __restrict__ w1,
    float* __restrict__ low) {
  __shared__ float part[16][17];
  const int t = blockIdx.x;
  const int r = threadIdx.x & 15, seg = threadIdx.x >> 4;
  const bf16* hr = h + (size_t)t * HM;
  float acc = 0.f;
  const int i0 = seg * 64;
  for (int i = i0; i < i0 + 64; ++i) acc += (float)hr[i] * w1[i * 16 + r];
  part[seg][r] = acc;
  __syncthreads();
  if (threadIdx.x < 16) {
    float s = 0.f;
    for (int j = 0; j < 16; ++j) s += part[j][threadIdx.x];
    low[(size_t)t * 16 + threadIdx.x] = s;
  }
}

// ---------------------------------------------------------------------------
// gk = log_sigmoid(low @ gk_w2 + b2) / 16    -> f32 [BT, 512]
// ---------------------------------------------------------------------------
__global__ __launch_bounds__(128) void gk_kernel(const float* __restrict__ low,
                                                 const float* __restrict__ w2,
                                                 const float* __restrict__ b2,
                                                 float* __restrict__ gk) {
  __shared__ float lo[16];
  const int t = blockIdx.x;
  if (threadIdx.x < 16) lo[threadIdx.x] = low[(size_t)t * 16 + threadIdx.x];
  __syncthreads();
  for (int c = threadIdx.x; c < HK; c += 128) {
    float a = b2[c];
#pragma unroll
    for (int i = 0; i < 16; ++i) a += lo[i] * w2[i * HK + c];
    float ls = fminf(a, 0.f) - log1pf(__expf(-fabsf(a)));
    gk[(size_t)t * HK + c] = ls * (1.0f / 16.0f);
  }
}

// ---------------------------------------------------------------------------
// WMMA GEMM: out[M,N] = A[M,K](bf16) @ BT[N,K](bf16)^T  (+ epilogue)
// One wave computes a 16(M) x 64(N) strip; 8 waves / block.
// ---------------------------------------------------------------------------
enum { MODE_BF16 = 0, MODE_QSCALE = 1, MODE_RESID = 2, MODE_GELU = 3,
       MODE_RESID_BIAS = 4 };

template <int MODE>
__global__ __launch_bounds__(256) void gemm_bf16_wmma(
    const bf16* __restrict__ A, const bf16* __restrict__ BTw,
    const float* __restrict__ bias, const float* __restrict__ resid,
    void* __restrict__ outp, int M, int N, int K) {
  const int lane = threadIdx.x;
  const int tilesN = N >> 6;
  const int tilesM = M >> 4;
  const int w = blockIdx.x * 8 + threadIdx.y;
  if (w >= tilesM * tilesN) return;
  const int tm = w / tilesN, tn = w % tilesN;
  const int rrow = lane & 15;
  const int akb = (lane >> 4) << 4;   // A-frag K offset (0/16)
  const int bkh = (lane >> 4) << 4;   // B-frag K half (0/16)
  const int ncol = lane & 15;

  const bf16* Arow = A + (size_t)(tm * 16 + rrow) * K;
  const bf16* Bc[4];
#pragma unroll
  for (int nt = 0; nt < 4; ++nt)
    Bc[nt] = BTw + (size_t)(tn * 64 + nt * 16 + ncol) * K;

  v8f acc[4];
#pragma unroll
  for (int nt = 0; nt < 4; ++nt) acc[nt] = v8f_zero();

  for (int k0 = 0; k0 < K; k0 += 32) {
    __builtin_prefetch(Arow + k0 + 128, 0, 1);
    v16bf a = make_afrag(Arow + k0, akb);
#pragma unroll
    for (int nt = 0; nt < 4; ++nt) {
      v16bf b = *(const v16bf*)(Bc[nt] + k0 + bkh);  // contiguous (transposed W)
      acc[nt] = wmma_bf16(a, b, acc[nt]);
    }
  }

#pragma unroll
  for (int nt = 0; nt < 4; ++nt) {
#pragma unroll
    for (int r = 0; r < 8; ++r) {
      const int orow = tm * 16 + r + ((lane >> 4) << 3);
      const int ocol = tn * 64 + nt * 16 + ncol;
      const size_t oi = (size_t)orow * N + ocol;
      const float val = acc[nt][r];
      if constexpr (MODE == MODE_QSCALE) {
        ((bf16*)outp)[oi] = (bf16)(val * 0.17677669529663687f);  // DK^-0.5
      } else if constexpr (MODE == MODE_BF16) {
        ((bf16*)outp)[oi] = (bf16)val;
      } else if constexpr (MODE == MODE_RESID) {
        ((float*)outp)[oi] = resid[oi] + val;
      } else if constexpr (MODE == MODE_GELU) {
        float t = val + bias[ocol];
        ((bf16*)outp)[oi] =
            (bf16)(0.5f * t * (1.0f + erff(t * 0.70710678118654752f)));
      } else {  // MODE_RESID_BIAS
        ((float*)outp)[oi] = resid[oi] + val + bias[ocol];
      }
    }
  }
}

// ---------------------------------------------------------------------------
// Chunked GLA recurrence, one wave per (batch, head).
//  per chunk (C=64):  b_t = cumsum(gk) ; qe = q*e^b ; kd = k*e^-b ;
//   A = tril(qe kd^T) ; o = qe S + A v ; S = diag(e^B)(S + kd^T v)
//   epilogue: RMSNorm(o)*rms_w * swish(g) -> og (bf16)
// V tiles are DMA'd into double-buffered LDS by the Tensor Data Mover
// (TENSORcnt), overlapping the previous chunk's WMMA work.
// ---------------------------------------------------------------------------
__global__ __launch_bounds__(32) void gla_chunk_kernel(
    const bf16* __restrict__ qg, const bf16* __restrict__ kg,
    const bf16* __restrict__ vg, const float* __restrict__ gkg,
    const bf16* __restrict__ gg, const float* __restrict__ rmsw,
    bf16* __restrict__ og) {
  const int lane = threadIdx.x;          // = state k index (0..31)
  const int b = blockIdx.x >> 4;
  const int hh = blockIdx.x & 15;

  __shared__ alignas(32) bf16 s_qe[CH * DK];
  __shared__ alignas(32) bf16 s_kd[CH * DK];
  __shared__ alignas(32) bf16 s_v[2][CH * DV];
  __shared__ alignas(32) bf16 s_A[CH * CH];
  __shared__ alignas(32) bf16 s_S[DK * DV];
  __shared__ float s_o[CH * DV];
  __shared__ float s_eB[DK];
  __shared__ float s_rw[DV];

  s_rw[lane] = rmsw[lane];
  s_rw[lane + 32] = rmsw[lane + 32];

  v8f S[2][4];
#pragma unroll
  for (int mt = 0; mt < 2; ++mt)
#pragma unroll
    for (int nt = 0; nt < 4; ++nt) S[mt][nt] = v8f_zero();

  const int rrow = lane & 15;
  const int hi = lane >> 4;
  const int akb = hi << 4;
  const int bkh = hi << 4;
  const int nn = lane & 15;

  const size_t vhead = ((size_t)b * TT) * HV + hh * DV;  // + t*HV

#if GLA_USE_TDM
  tdm_load_v_tile(vg + vhead, &s_v[0][0]);  // prefetch chunk 0
#endif

  for (int c = 0; c < NCH; ++c) {
    const int t0 = c * CH;
    const int cur = c & 1;
    const size_t qbase = ((size_t)b * TT + t0) * HK + hh * DK + lane;
    const size_t vbase = vhead + (size_t)t0 * HV;

    // ---- phase 1: gating prep (lane = k index) + stage S_prev ----
    float bl = 0.f;
    for (int t = 0; t < CH; ++t) {
      const size_t ridx = qbase + (size_t)t * HK;
      bl += gkg[ridx];
      s_qe[t * DK + lane] = (bf16)((float)qg[ridx] * __expf(bl));
      s_kd[t * DK + lane] = (bf16)((float)kg[ridx] * __expf(-bl));
    }
    s_eB[lane] = __expf(bl);
#if !GLA_USE_TDM
    for (int t = lane; t < CH; t += 32) {
      const bf16* src = vg + vbase + (size_t)t * HV;
#pragma unroll
      for (int j = 0; j < 8; ++j)
        *(v8bf*)(&s_v[cur][t * DV + j * 8]) = *(const v8bf*)(src + j * 8);
    }
#endif
    {  // stage S_prev (D-layout accumulators -> LDS row-major bf16)
#pragma unroll
      for (int mt = 0; mt < 2; ++mt)
#pragma unroll
        for (int nt = 0; nt < 4; ++nt)
#pragma unroll
          for (int r = 0; r < 8; ++r)
            s_S[(mt * 16 + r + (hi << 3)) * DV + nt * 16 + nn] =
                (bf16)S[mt][nt][r];
    }
#if GLA_USE_TDM
    __builtin_amdgcn_s_wait_tensorcnt(0);  // v tile for this chunk landed
#endif
    __syncthreads();

#if GLA_USE_TDM
    if (c + 1 < NCH)  // overlap next chunk's DMA with this chunk's WMMAs
      tdm_load_v_tile(vg + vhead + (size_t)(t0 + CH) * HV, &s_v[cur ^ 1][0]);
#endif

    // hoisted B-fragments of v and S_prev (valid through phases 2-4)
    v16bf bS[4], bV0[4], bV1[4], bkd[4];
#pragma unroll
    for (int nt = 0; nt < 4; ++nt) {
      bS[nt] = lds_bfrag(s_S, DV, bkh, nt * 16 + nn);
      bV0[nt] = lds_bfrag(&s_v[cur][0], DV, bkh, nt * 16 + nn);
      bV1[nt] = lds_bfrag(&s_v[cur][0], DV, 32 + bkh, nt * 16 + nn);
      // kd^T B-frag: e_i = kd[s][k] with s = column -> contiguous row of s_kd
      bkd[nt] = *(const v16bf*)(&s_kd[(nt * 16 + nn) * DK + bkh]);
    }

    // ---- phase 2: A = tril(qe @ kd^T) -> s_A (bf16) ----
#pragma unroll
    for (int mt = 0; mt < 4; ++mt) {
      v16bf aq = make_afrag(&s_qe[(mt * 16 + rrow) * DK], akb);
#pragma unroll
      for (int nt = 0; nt < 4; ++nt) {
        v8f a0 = wmma_bf16(aq, bkd[nt], v8f_zero());
#pragma unroll
        for (int r = 0; r < 8; ++r) {
          const int m = mt * 16 + r + (hi << 3);
          const int n = nt * 16 + nn;
          s_A[m * CH + n] = (bf16)((n <= m) ? a0[r] : 0.0f);
        }
      }
    }
    __syncthreads();

    // ---- phase 3: o = qe @ S_prev + A @ v -> s_o (f32) ----
#pragma unroll
    for (int mt = 0; mt < 4; ++mt) {
      v16bf aq = make_afrag(&s_qe[(mt * 16 + rrow) * DK], akb);
      v16bf aA0 = make_afrag(&s_A[(mt * 16 + rrow) * CH], akb);
      v16bf aA1 = make_afrag(&s_A[(mt * 16 + rrow) * CH], 32 + akb);
#pragma unroll
      for (int nt = 0; nt < 4; ++nt) {
        v8f acc = wmma_bf16(aq, bS[nt], v8f_zero());
        acc = wmma_bf16(aA0, bV0[nt], acc);
        acc = wmma_bf16(aA1, bV1[nt], acc);
#pragma unroll
        for (int r = 0; r < 8; ++r)
          s_o[(mt * 16 + r + (hi << 3)) * DV + nt * 16 + nn] = acc[r];
      }
    }

    // ---- phase 4: S = diag(e^B) * (S + kd^T @ v) ----
#pragma unroll
    for (int mt = 0; mt < 2; ++mt) {
      v16bf f0 = lds_afragT(s_kd, DK, akb, mt * 16 + rrow);
      v16bf f1 = lds_afragT(s_kd, DK, 32 + akb, mt * 16 + rrow);
#pragma unroll
      for (int nt = 0; nt < 4; ++nt) {
        v8f acc = wmma_bf16(f0, bV0[nt], S[mt][nt]);
        acc = wmma_bf16(f1, bV1[nt], acc);
#pragma unroll
        for (int r = 0; r < 8; ++r)
          acc[r] *= s_eB[mt * 16 + r + (hi << 3)];
        S[mt][nt] = acc;
      }
    }
    __syncthreads();

    // ---- phase 5: og = RMSNorm(o)*rms_w * g*sigmoid(g) ----
    for (int t = lane; t < CH; t += 32) {
      float ssq = 0.f;
#pragma unroll
      for (int j = 0; j < DV; ++j) {
        float v = s_o[t * DV + j];
        ssq += v * v;
      }
      const float inv = rsqrtf(ssq * (1.0f / DV) + 1e-5f);
      const bf16* gr = gg + vbase + (size_t)t * HV;
      bf16* outr = og + vbase + (size_t)t * HV;
#pragma unroll
      for (int j = 0; j < DV; ++j) {
        float o = s_o[t * DV + j] * inv * s_rw[j];
        float gvv = (float)gr[j];
        float sw = gvv / (1.0f + __expf(-gvv));  // g*sigmoid(g)
        outr[j] = (bf16)(o * sw);
      }
    }
    __syncthreads();
  }
}

// ---------------------------------------------------------------------------
// Workspace layout (bytes; all sizes multiples of 256)
// ---------------------------------------------------------------------------
constexpr size_t OFF_WQT = 0;
constexpr size_t OFF_WKT = OFF_WQT + (size_t)HK * HM * 2;
constexpr size_t OFF_WVT = OFF_WKT + (size_t)HK * HM * 2;
constexpr size_t OFF_WGT = OFF_WVT + (size_t)HV * HM * 2;
constexpr size_t OFF_WOT = OFF_WGT + (size_t)HV * HM * 2;
constexpr size_t OFF_W1T = OFF_WOT + (size_t)HM * HV * 2;
constexpr size_t OFF_W2T = OFF_W1T + (size_t)D2 * HM * 2;
constexpr size_t OFF_H   = OFF_W2T + (size_t)HM * D2 * 2;
constexpr size_t OFF_Q   = OFF_H  + (size_t)BT * HM * 2;
constexpr size_t OFF_K   = OFF_Q  + (size_t)BT * HK * 2;
constexpr size_t OFF_V   = OFF_K  + (size_t)BT * HK * 2;
constexpr size_t OFF_G   = OFF_V  + (size_t)BT * HV * 2;
constexpr size_t OFF_GK  = OFF_G  + (size_t)BT * HV * 2;
constexpr size_t OFF_LOW = OFF_GK + (size_t)BT * HK * 4;
constexpr size_t OFF_OG  = OFF_LOW + (size_t)BT * 16 * 4;
constexpr size_t OFF_XM  = OFF_OG + (size_t)BT * HV * 2;
constexpr size_t OFF_H2  = OFF_V;   // v dead after recurrence
constexpr size_t OFF_MID = OFF_H;   // h,q,k dead after recurrence (16MB region)

static inline int gemm_blocks(int M, int N) {
  return ((M >> 4) * (N >> 6) + 7) / 8;
}

extern "C" void kernel_launch(void* const* d_in, const int* in_sizes, int n_in,
                              void* d_out, int out_size, void* d_ws,
                              size_t ws_size, hipStream_t stream) {
  (void)in_sizes; (void)n_in; (void)out_size; (void)ws_size;
  const float* x     = (const float*)d_in[0];
  const float* ln_w  = (const float*)d_in[1];
  const float* ln_b  = (const float*)d_in[2];
  const float* Wq    = (const float*)d_in[3];
  const float* Wk    = (const float*)d_in[4];
  const float* Wv    = (const float*)d_in[5];
  const float* gk_w1 = (const float*)d_in[6];
  const float* gk_w2 = (const float*)d_in[7];
  const float* gk_b2 = (const float*)d_in[8];
  const float* Wg    = (const float*)d_in[9];
  const float* rms_w = (const float*)d_in[10];
  const float* Wo    = (const float*)d_in[11];
  const float* W1    = (const float*)d_in[12];
  const float* b1    = (const float*)d_in[13];
  const float* W2    = (const float*)d_in[14];
  const float* b2    = (const float*)d_in[15];
  float* out = (float*)d_out;

  char* ws = (char*)d_ws;
  bf16* WqT = (bf16*)(ws + OFF_WQT);
  bf16* WkT = (bf16*)(ws + OFF_WKT);
  bf16* WvT = (bf16*)(ws + OFF_WVT);
  bf16* WgT = (bf16*)(ws + OFF_WGT);
  bf16* WoT = (bf16*)(ws + OFF_WOT);
  bf16* W1T = (bf16*)(ws + OFF_W1T);
  bf16* W2T = (bf16*)(ws + OFF_W2T);
  bf16* h    = (bf16*)(ws + OFF_H);
  bf16* qb   = (bf16*)(ws + OFF_Q);
  bf16* kb   = (bf16*)(ws + OFF_K);
  bf16* vb   = (bf16*)(ws + OFF_V);
  bf16* gb   = (bf16*)(ws + OFF_G);
  float* gk  = (float*)(ws + OFF_GK);
  float* low = (float*)(ws + OFF_LOW);
  bf16* ogb  = (bf16*)(ws + OFF_OG);
  float* xm  = (float*)(ws + OFF_XM);
  bf16* h2   = (bf16*)(ws + OFF_H2);
  bf16* mid  = (bf16*)(ws + OFF_MID);

  const dim3 gblk(32, 8, 1);

  // 1) weight transposes -> bf16 [N,K]
  transpose_bf16_kernel<<<(HM * HK) / 256, 256, 0, stream>>>(Wq, WqT, HM, HK);
  transpose_bf16_kernel<<<(HM * HK) / 256, 256, 0, stream>>>(Wk, WkT, HM, HK);
  transpose_bf16_kernel<<<(HM * HV) / 256, 256, 0, stream>>>(Wv, WvT, HM, HV);
  transpose_bf16_kernel<<<(HM * HV) / 256, 256, 0, stream>>>(Wg, WgT, HM, HV);
  transpose_bf16_kernel<<<(HV * HM) / 256, 256, 0, stream>>>(Wo, WoT, HV, HM);
  transpose_bf16_kernel<<<(HM * D2) / 256, 256, 0, stream>>>(W1, W1T, HM, D2);
  transpose_bf16_kernel<<<(D2 * HM) / 256, 256, 0, stream>>>(W2, W2T, D2, HM);

  // 2) h = LN(x) -> bf16
  ln_bf16_kernel<<<BT, 256, 0, stream>>>(x, ln_w, ln_b, h);

  // 3) projections (WMMA)
  gemm_bf16_wmma<MODE_QSCALE><<<gemm_blocks(BT, HK), gblk, 0, stream>>>(
      h, WqT, nullptr, nullptr, (void*)qb, BT, HK, HM);
  gemm_bf16_wmma<MODE_BF16><<<gemm_blocks(BT, HK), gblk, 0, stream>>>(
      h, WkT, nullptr, nullptr, (void*)kb, BT, HK, HM);
  gemm_bf16_wmma<MODE_BF16><<<gemm_blocks(BT, HV), gblk, 0, stream>>>(
      h, WvT, nullptr, nullptr, (void*)vb, BT, HV, HM);
  gemm_bf16_wmma<MODE_BF16><<<gemm_blocks(BT, HV), gblk, 0, stream>>>(
      h, WgT, nullptr, nullptr, (void*)gb, BT, HV, HM);

  // 4) low-rank gate -> gk (f32)
  lowrank_kernel<<<BT, 256, 0, stream>>>(h, gk_w1, low);
  gk_kernel<<<BT, 128, 0, stream>>>(low, gk_w2, gk_b2, gk);

  // 5) chunked GLA recurrence + RMSNorm*swish gate -> og (bf16)
  gla_chunk_kernel<<<BB * HEADS, 32, 0, stream>>>(qb, kb, vb, gk, gb, rms_w,
                                                  ogb);

  // 6) x_mid = x + og @ Wo
  gemm_bf16_wmma<MODE_RESID><<<gemm_blocks(BT, HM), gblk, 0, stream>>>(
      ogb, WoT, nullptr, x, (void*)xm, BT, HM, HV);

  // 7) h2 = LN(x_mid)
  ln_bf16_kernel<<<BT, 256, 0, stream>>>(xm, ln_w, ln_b, h2);

  // 8) mid = gelu(h2 @ W1 + b1) -> bf16
  gemm_bf16_wmma<MODE_GELU><<<gemm_blocks(BT, D2), gblk, 0, stream>>>(
      h2, W1T, b1, nullptr, (void*)mid, BT, D2, HM);

  // 9) out = x_mid + mid @ W2 + b2
  gemm_bf16_wmma<MODE_RESID_BIAS><<<gemm_blocks(BT, HM), gblk, 0, stream>>>(
      mid, W2T, b2, xm, (void*)out, BT, HM, D2);
}
